// OrthoClassDynProjectSimplexLips_78469052498206
// MI455X (gfx1250) — matrix operand order repaired and checked by reading, not compiled, not run on hardware
//
#include <hip/hip_runtime.h>

// ---------------------------------------------------------------------------
// Types for gfx1250 WMMA (wave32)
// ---------------------------------------------------------------------------
typedef __attribute__((ext_vector_type(16))) __bf16 v16bf;
typedef __attribute__((ext_vector_type(8)))  __bf16 bf16x8;
typedef __attribute__((ext_vector_type(8)))  float  v8f;
typedef __attribute__((ext_vector_type(4)))  int    i32x4;
typedef __attribute__((ext_vector_type(8)))  int    i32x8;

#define NS_ITERS   25     // Newton-Schulz iterations for inv(I+A)
#define RED_BLOCKS 512    // blocks for deterministic two-stage reductions

#define KC      64        // K-chunk staged per TDM transfer
#define BROWS   64        // B rows (N) per block tile
#define LDSROW  72        // LDS row pitch in bf16 elems: 64 + 8 pad (TDM pad 16B/128B)

// ---------------------------------------------------------------------------
// Helpers
// ---------------------------------------------------------------------------
__device__ __forceinline__ __bf16 f2bf(float f) {
  unsigned int u = __float_as_uint(f);
  unsigned int r = (u + 0x7FFFu + ((u >> 16) & 1u)) >> 16;   // RNE
  union { unsigned short s; __bf16 b; } o; o.s = (unsigned short)r;
  return o.b;
}
__device__ __forceinline__ void storev(float*  p, float v) { *p = v; }
__device__ __forceinline__ void storev(__bf16* p, float v) { *p = f2bf(v); }

__device__ __forceinline__ v8f wmma_bf16(v16bf a, v16bf b, v8f c) {
  return __builtin_amdgcn_wmma_f32_16x16x32_bf16(
      /*neg_a=*/false, a, /*neg_b=*/false, b,
      /*c_mod=*/(short)0, c, /*reuse_a=*/false, /*reuse_b=*/false);
}

// A fragment (16x32, row m per lane, K split 8|8 per lane-half), from global
__device__ __forceinline__ v16bf load_frag_a(const __bf16* base, int ld, int row, int k0) {
  const int lane = threadIdx.x & 31;
  const int m    = lane & 15;
  const int kh   = (lane >> 4) << 3;   // 0 or 8
  const __bf16* p = base + (size_t)(row + m) * (size_t)ld + (size_t)(k0 + kh);
  bf16x8 lo = *(const bf16x8*)(p);
  bf16x8 hi = *(const bf16x8*)(p + 16);
  v16bf f;
#pragma unroll
  for (int i = 0; i < 8; ++i) { f[i] = lo[i]; f[i + 8] = hi[i]; }
  return f;
}

// B fragment (32x16, column n per lane, K split 8|8 per lane-half, same as A),
// read from the TDM-staged LDS tile (row pitch LDSROW, conflict-free).
__device__ __forceinline__ v16bf lds_frag_b(const __bf16* Bs, int col_local, int k_local) {
  const int lane = threadIdx.x & 31;
  const int n    = lane & 15;
  const int kh   = (lane >> 4) << 3;   // 0 or 8
  const __bf16* p = Bs + (size_t)(col_local + n) * LDSROW + (size_t)(k_local + kh);
  bf16x8 lo = *(const bf16x8*)(p);
  bf16x8 hi = *(const bf16x8*)(p + 16);
  v16bf f;
#pragma unroll
  for (int i = 0; i < 8; ++i) { f[i] = lo[i]; f[i + 8] = hi[i]; }
  return f;
}

// ---------------------------------------------------------------------------
// Tensor Data Mover: load one 64(rows) x 64(K) bf16 tile into LDS.
// D# per CDNA5 ISA: group0 (4 SGPRs) + group1 (8 SGPRs),
// 2-D tile, data_size=2B, LDS padding 4 DWORDs per 32 DWORDs (-> 144B pitch).
// ---------------------------------------------------------------------------
__device__ __forceinline__ void tdm_load_tile(const __bf16* gsrc, unsigned lds_off, int ldb) {
  unsigned long long ga = (unsigned long long)gsrc & ((1ull << 57) - 1ull);
  const unsigned TD0 = 1u << 20, TD1 = 1u << 20;   // generous tensor dims (tiles never OOB)
  i32x4 g0;
  g0[0] = 1;                                             // count=1, user mode, no gather
  g0[1] = (int)lds_off;                                  // lds_addr
  g0[2] = (int)(unsigned)(ga & 0xFFFFFFFFull);           // global_addr[31:0]
  g0[3] = (int)((unsigned)(ga >> 32) | (2u << 30));      // global_addr[56:32] | type=2
  i32x8 g1;
  g1[0] = (int)((1u << 16)      // data_size = 2B
              | (1u << 20)      // pad_enable
              | (4u << 22)      // pad_interval: every 32 DWORDs (=128B row)
              | (3u << 25));    // pad_amount: 4 DWORDs (=16B)
  g1[1] = (int)((TD0 & 0xFFFFu) << 16);                            // tensor_dim0 lo
  g1[2] = (int)(((TD0 >> 16) & 0xFFFFu) | ((TD1 & 0xFFFFu) << 16));// dim0 hi | dim1 lo
  g1[3] = (int)(((TD1 >> 16) & 0xFFFFu) | ((unsigned)KC << 16));   // dim1 hi | tile_dim0=64
  g1[4] = (int)BROWS;                                              // tile_dim1=64, tile_dim2=0
  g1[5] = (int)(unsigned)ldb;                                      // tensor_dim0_stride lo32
  g1[6] = 0;                                                       // stride hi | dim1_stride lo
  g1[7] = 0;
  asm volatile("tensor_load_to_lds %0, %1" :: "s"(g0), "s"(g1) : "memory");
}

__device__ __forceinline__ void wait_tensor0() {
#if __has_builtin(__builtin_amdgcn_s_wait_tensorcnt)
  __builtin_amdgcn_s_wait_tensorcnt(0);
#else
  asm volatile("s_wait_tensorcnt 0x0" ::: "memory");
#endif
}
__device__ __forceinline__ void wait_tensor1() {
#if __has_builtin(__builtin_amdgcn_s_wait_tensorcnt)
  __builtin_amdgcn_s_wait_tensorcnt(1);
#else
  asm volatile("s_wait_tensorcnt 0x1" ::: "memory");
#endif
}

// ---------------------------------------------------------------------------
// Canonical TN GEMM:  C(MxN) = alpha * (A(MxK) * B'(NxK)^T) [+ A2*B2'^T] + ...
// B' rows staged to LDS by the TDM (double buffered), A direct from global.
// Invariant used: ldb == K for every B operand in this pipeline.
// Block: 256 threads = 8 waves (4x2), wave tile 32x32 (2x2 WMMA frags).
// Requires M%128==0, N%64==0, K%64==0 (true for every GEMM here).
// ---------------------------------------------------------------------------
__global__ void __launch_bounds__(256)
gemm_tn_k(const __bf16* __restrict__ A,  int lda,  const __bf16* __restrict__ B,  int K,
          const __bf16* __restrict__ A2, int lda2, const __bf16* __restrict__ B2, int K2,
          float* __restrict__ Cf, __bf16* __restrict__ Cb, int ldc,
          float alpha, float diag, int relu,
          const float* __restrict__ bias1, const float* __restrict__ bias2)
{
  __shared__ __bf16 Bs[2][BROWS * LDSROW];
  const int wave = threadIdx.x >> 5;
  const int row  = blockIdx.y * 128 + (wave & 3) * 32;
  const int colw = (wave >> 2) * 32;            // column offset inside the 64-wide tile
  const int col0 = blockIdx.x * 64;             // B-row (N) base of this block
  const bool w0  = (wave == 0);
  const unsigned lds0 = (unsigned)(unsigned long long)&Bs[0][0];
  const unsigned lds1 = (unsigned)(unsigned long long)&Bs[1][0];

  v8f c00 = {}, c01 = {}, c10 = {}, c11 = {};

  auto run = [&](const __bf16* Ax, int ldax, const __bf16* Bx, int Kx) {
    const __bf16* gB = Bx + (size_t)col0 * (size_t)Kx;    // ldb == Kx
    const int nch = Kx / KC;
    if (w0) tdm_load_tile(gB, lds0, Kx);                  // preload chunk 0
    for (int ic = 0; ic < nch; ++ic) {
      const int buf = ic & 1;
      const bool more = (ic + 1 < nch);
      if (w0) {
        if (more) tdm_load_tile(gB + (size_t)(ic + 1) * KC, buf ? lds0 : lds1, Kx);
        if (more) wait_tensor1(); else wait_tensor0();    // current buffer resident
      }
      __syncthreads();                                    // publish LDS tile
      const __bf16* Bt = Bs[buf];
      const int kc = ic * KC;
#pragma unroll
      for (int ks = 0; ks < KC; ks += 32) {
        v16bf a0 = load_frag_a(Ax, ldax, row,      kc + ks);
        v16bf a1 = load_frag_a(Ax, ldax, row + 16, kc + ks);
        v16bf b0 = lds_frag_b(Bt, colw,      ks);
        v16bf b1 = lds_frag_b(Bt, colw + 16, ks);
        c00 = wmma_bf16(a0, b0, c00);
        c01 = wmma_bf16(a0, b1, c01);
        c10 = wmma_bf16(a1, b0, c10);
        c11 = wmma_bf16(a1, b1, c11);
      }
      __syncthreads();                                    // tile consumed; safe to overwrite
    }
  };
  run(A, lda, B, K);
  if (A2) run(A2, lda2, B2, K2);

  const int lane = threadIdx.x & 31;
  const int nl   = lane & 15;
  const int hi   = lane >> 4;
  auto emit = [&](const v8f& c, int rb, int cb) {
#pragma unroll
    for (int r = 0; r < 8; ++r) {
      const int m = row + rb + r + 8 * hi;   // C/D layout: VGPR r -> M=r (+8 hi half)
      const int n = col0 + colw + cb + nl;
      float v = alpha * c[r];
      if (bias1) v += bias1[n];
      if (bias2) v += bias2[n];
      if (m == n) v += diag;
      if (relu) v = fmaxf(v, 0.f);
      if (Cf) Cf[(size_t)m * ldc + n] = v;
      else    Cb[(size_t)m * ldc + n] = f2bf(v);
    }
  };
  emit(c00, 0, 0); emit(c01, 0, 16); emit(c10, 16, 0); emit(c11, 16, 16);
}

// ---------------------------------------------------------------------------
// Elementwise / transpose / reduction utility kernels
// ---------------------------------------------------------------------------
__global__ void zero_k(float* p, int n) {
  int i = blockIdx.x * blockDim.x + threadIdx.x;
  if (i < n) p[i] = 0.f;
}

__global__ void convert_bf16_k(__bf16* __restrict__ dst, const float* __restrict__ src,
                               size_t n, const float* __restrict__ sptr) {
  const float s = sptr ? sptr[0] : 1.f;
  for (size_t i = (size_t)blockIdx.x * blockDim.x + threadIdx.x; i < n;
       i += (size_t)gridDim.x * blockDim.x)
    dst[i] = f2bf(src[i] * s);
}

// dst (CxR) = sconst*(sptr?*sptr:1) * src(RxC)^T  + diag on the main diagonal
template <typename OutT>
__global__ void transpose_scale_k(OutT* __restrict__ dst, const float* __restrict__ src,
                                  int R, int C, const float* __restrict__ sptr,
                                  float sconst, float diag) {
  __shared__ float tile[32][33];
  const float s = sconst * (sptr ? sptr[0] : 1.f);
  const int i0 = blockIdx.y * 32, j0 = blockIdx.x * 32;
  const int tx = threadIdx.x, ty = threadIdx.y;   // 32 x 8
#pragma unroll
  for (int rr = 0; rr < 32; rr += 8) {
    int i = i0 + ty + rr, j = j0 + tx;
    tile[ty + rr][tx] = (i < R && j < C) ? src[(size_t)i * C + j] : 0.f;
  }
  __syncthreads();
#pragma unroll
  for (int rr = 0; rr < 32; rr += 8) {
    int j = j0 + ty + rr, i = i0 + tx;
    if (j < C && i < R) {
      float v = s * tile[tx][ty + rr] + ((i == j) ? diag : 0.f);
      storev(&dst[(size_t)j * R + i], v);
    }
  }
}

// M(cxc) = s*U - s*U^T + G + I, U = top cxc block of W (row-major, ld=c)
__global__ void build_M_k(float* __restrict__ Mo, const float* __restrict__ W, int c,
                          const float* __restrict__ G, const float* __restrict__ sptr) {
  int j = blockIdx.x * blockDim.x + threadIdx.x;
  int i = blockIdx.y;
  if (j >= c) return;
  const float s = sptr[0];
  float v = s * W[(size_t)i * c + j] - s * W[(size_t)j * c + i];
  if (G) v += G[(size_t)i * c + j];
  if (i == j) v += 1.f;
  Mo[(size_t)i * c + j] = v;
}

__global__ void ssq_partial_k(const float* __restrict__ w, size_t n, float* __restrict__ part) {
  __shared__ float red[256];
  float s = 0.f;
  for (size_t i = (size_t)blockIdx.x * blockDim.x + threadIdx.x; i < n;
       i += (size_t)gridDim.x * blockDim.x) {
    float v = w[i]; s += v * v;
  }
  red[threadIdx.x] = s; __syncthreads();
  for (int o = 128; o > 0; o >>= 1) {
    if ((int)threadIdx.x < o) red[threadIdx.x] += red[threadIdx.x + o];
    __syncthreads();
  }
  if (threadIdx.x == 0) part[blockIdx.x] = red[0];
}

// s_out = alpha / sqrt(sum(part))
__global__ void ssq_final_k(const float* __restrict__ part, int nb,
                            const float* __restrict__ alpha, float* __restrict__ s_out) {
  __shared__ float red[256];
  float s = 0.f;
  for (int i = threadIdx.x; i < nb; i += blockDim.x) s += part[i];
  red[threadIdx.x] = s; __syncthreads();
  for (int o = 128; o > 0; o >>= 1) {
    if ((int)threadIdx.x < o) red[threadIdx.x] += red[threadIdx.x + o];
    __syncthreads();
  }
  if (threadIdx.x == 0) s_out[0] = alpha[0] * rsqrtf(red[0]);
}

// out = max over blocks of sum_j |M[b*os + j*is]|   (atomicMax, order-independent)
__global__ void abs_sum_max_k(const float* __restrict__ M, int n, int os, int is,
                              float* __restrict__ out) {
  __shared__ float red[256];
  const int b = blockIdx.x;
  float s = 0.f;
  for (int j = threadIdx.x; j < n; j += blockDim.x)
    s += fabsf(M[(size_t)b * os + (size_t)j * is]);
  red[threadIdx.x] = s; __syncthreads();
  for (int o = 128; o > 0; o >>= 1) {
    if ((int)threadIdx.x < o) red[threadIdx.x] += red[threadIdx.x + o];
    __syncthreads();
  }
  if (threadIdx.x == 0) atomicMax((int*)out, __float_as_int(red[0]));
}

__global__ void inv_prod_k(const float* a, const float* b, float* o) {
  if (threadIdx.x == 0 && blockIdx.x == 0) o[0] = 1.f / (a[0] * b[0]);
}

// ---------------------------------------------------------------------------
// Projection: per-row (128 cols) 30-step Newton on lambda; one wave32 per row.
// ---------------------------------------------------------------------------
__global__ void project_k(const float* __restrict__ ft, const float* __restrict__ h,
                          float* __restrict__ out, int rows) {
  const int row  = blockIdx.x * (blockDim.x >> 5) + (threadIdx.x >> 5);
  const int lane = threadIdx.x & 31;
  if (row >= rows) return;
  const float* fr = ft + (size_t)row * 128;
  const float* hr = h  + (size_t)row * 128;
  float f[4], lo[4], s = 0.f;
#pragma unroll
  for (int j = 0; j < 4; ++j) {
    f[j]  = fr[lane + 32 * j];
    lo[j] = -100.f * (expf(0.02f * hr[lane + 32 * j]) - 1.f);   // ALPHA_1, SIGMA_1
    s += f[j];
  }
  for (int o = 16; o > 0; o >>= 1) s += __shfl_xor(s, o, 32);
  float lam = s * (1.f / 128.f);
  for (int it = 0; it < 30; ++it) {
    float g = 0.f, a = 0.f;
#pragma unroll
    for (int j = 0; j < 4; ++j) {
      float d = f[j] - lam;
      g += fmaxf(d, lo[j]);
      a += (d > lo[j]) ? 1.f : 0.f;
    }
    for (int o = 16; o > 0; o >>= 1) { g += __shfl_xor(g, o, 32); a += __shfl_xor(a, o, 32); }
    lam += g / fmaxf(a, 1.f);
  }
  float* orow = out + (size_t)row * 128;
#pragma unroll
  for (int j = 0; j < 4; ++j) orow[lane + 32 * j] = fmaxf(f[j] - lam, lo[j]);
}

// ---------------------------------------------------------------------------
// Host orchestration
// ---------------------------------------------------------------------------
extern "C" void kernel_launch(void* const* d_in, const int* in_sizes, int n_in,
                              void* d_out, int out_size, void* d_ws, size_t ws_size,
                              hipStream_t stream) {
  (void)in_sizes; (void)n_in; (void)out_size; (void)ws_size;
  const int Bn = 8192, NH = 128, Md = 4096, XD = 1024;

  const float* h    = (const float*)d_in[0];
  const float* x    = (const float*)d_in[1];
  const float* W_hm = (const float*)d_in[2];
  const float* b_hm = (const float*)d_in[3];
  const float* a_hm = (const float*)d_in[4];
  const float* W_mm = (const float*)d_in[5];
  const float* b_mm = (const float*)d_in[6];
  const float* a_mm = (const float*)d_in[7];
  const float* W_mh = (const float*)d_in[8];
  const float* b_mh = (const float*)d_in[9];
  const float* a_mh = (const float*)d_in[10];
  const float* W_ux = (const float*)d_in[11];
  const float* b_ux = (const float*)d_in[12];
  const float* a_ux = (const float*)d_in[13];

  // ---- workspace bump allocator (256B aligned) ----
  size_t off = 0;
  auto alloc = [&](size_t bytes) -> void* {
    void* p = (char*)d_ws + off;
    off += (bytes + 255) & ~(size_t)255;
    return p;
  };
  float*  part  = (float*)alloc(RED_BLOCKS * 4);
  float*  scal  = (float*)alloc(256);             // [0]=s [1]=r1 [2]=rinf [3]=x0scale
  float*  sval  = scal + 0;
  float*  r1    = scal + 1;
  float*  x0s   = scal + 3;
  float*  Mmat  = (float*)alloc((size_t)Md * Md * 4);
  float*  X     = (float*)alloc((size_t)Md * Md * 4);
  float*  T     = (float*)alloc((size_t)Md * Md * 4);
  float*  G     = (float*)alloc((size_t)XD * XD * 4);
  __bf16* Mb    = (__bf16*)alloc((size_t)Md * Md * 2);
  __bf16* Xb    = (__bf16*)alloc((size_t)Md * Md * 2);
  __bf16* XbT   = (__bf16*)alloc((size_t)Md * Md * 2);
  __bf16* TtB   = (__bf16*)alloc((size_t)Md * Md * 2);
  __bf16* Btop  = (__bf16*)alloc((size_t)Md * Md * 2);
  __bf16* Vb    = (__bf16*)alloc((size_t)3072 * 1024 * 2);
  __bf16* VbT   = (__bf16*)alloc((size_t)3072 * 1024 * 2);
  float*  Wt    = (float*)alloc((size_t)Md * NH * 4);
  float*  Qt    = (float*)alloc((size_t)Md * NH * 4);
  __bf16* Qb_hm = (__bf16*)alloc((size_t)Md * NH * 2);
  __bf16* Qb_ux = (__bf16*)alloc((size_t)Md * XD * 2);
  __bf16* Qb_mm = (__bf16*)alloc((size_t)Md * Md * 2);
  __bf16* Bf_mh = (__bf16*)alloc((size_t)NH * Md * 2);
  __bf16* hb    = (__bf16*)alloc((size_t)Bn * NH * 2);
  __bf16* xb    = (__bf16*)alloc((size_t)Bn * XD * 2);
  __bf16* zb    = (__bf16*)alloc((size_t)Bn * Md * 2);
  __bf16* z2b   = (__bf16*)alloc((size_t)Bn * Md * 2);
  float*  ftb   = (float*)alloc((size_t)Bn * NH * 4);

  // ---- launch helpers ----
  auto conv = [&](__bf16* dst, const float* src, size_t n, const float* sp) {
    size_t blocks = (n + 255) / 256; if (blocks > 8192) blocks = 8192;
    convert_bf16_k<<<(unsigned)blocks, 256, 0, stream>>>(dst, src, n, sp);
  };
  auto trp_f = [&](float* dst, const float* src, int R, int C,
                   const float* sp, float sc, float dg) {
    dim3 g((C + 31) / 32, (R + 31) / 32), b(32, 8);
    transpose_scale_k<float><<<g, b, 0, stream>>>(dst, src, R, C, sp, sc, dg);
  };
  auto trp_b = [&](__bf16* dst, const float* src, int R, int C,
                   const float* sp, float sc, float dg) {
    dim3 g((C + 31) / 32, (R + 31) / 32), b(32, 8);
    transpose_scale_k<__bf16><<<g, b, 0, stream>>>(dst, src, R, C, sp, sc, dg);
  };
  // B operands always N x K row-major with ldb == K (TDM descriptor relies on it)
  auto gemm = [&](const __bf16* A, int lda, const __bf16* Bm, int K,
                  float* Cf, __bf16* Cb, int M, int N, int ldc,
                  float alpha, float diag, int relu,
                  const float* bs1, const float* bs2,
                  const __bf16* A2 = nullptr, int lda2 = 0,
                  const __bf16* B2 = nullptr, int K2 = 0) {
    dim3 g(N / 64, M / 128), b(256);
    gemm_tn_k<<<g, b, 0, stream>>>(A, lda, Bm, K, A2, lda2, B2, K2,
                                   Cf, Cb, ldc, alpha, diag, relu, bs1, bs2);
  };

  // ---- Cayley orthogonalization: Q = [X(I-A) ; -2 V X], X = inv(I+A) via NS ----
  auto cayley_Q = [&](const float* W, int c, int cout, const float* alpha_ptr,
                      __bf16* Qb, float* Qf) {
    const size_t nW = (size_t)cout * c;
    const int r = cout - c;
    // s = alpha / ||W||_F
    ssq_partial_k<<<RED_BLOCKS, 256, 0, stream>>>(W, nW, part);
    ssq_final_k<<<1, 256, 0, stream>>>(part, RED_BLOCKS, alpha_ptr, sval);
    if (r > 0) {
      conv(Vb, W + (size_t)c * c, (size_t)r * c, sval);           // Vb = bf16(sV), r x c
      trp_b(VbT, W + (size_t)c * c, r, c, sval, 1.f, 0.f);        // VbT = bf16((sV)^T), c x r
      gemm(VbT, r, VbT, r, G, nullptr, c, c, c, 1.f, 0.f, 0, nullptr, nullptr); // G=(sV)^T(sV)
    }
    {
      dim3 g((c + 255) / 256, c);
      build_M_k<<<g, 256, 0, stream>>>(Mmat, W, c, (r > 0) ? G : nullptr, sval); // M = I + A
    }
    // Newton-Schulz init: X0 = M^T / (||M||_1 * ||M||_inf)
    zero_k<<<1, 64, 0, stream>>>(r1, 2);
    abs_sum_max_k<<<c, 256, 0, stream>>>(Mmat, c, c, 1, r1);
    abs_sum_max_k<<<c, 256, 0, stream>>>(Mmat, c, 1, c, r1 + 1);
    inv_prod_k<<<1, 1, 0, stream>>>(r1, r1 + 1, x0s);
    trp_f(X, Mmat, c, c, x0s, 1.f, 0.f);
    conv(Mb, Mmat, (size_t)c * c, nullptr);
    for (int it = 0; it < NS_ITERS; ++it) {
      trp_b(XbT, X, c, c, nullptr, 1.f, 0.f);
      conv(Xb, X, (size_t)c * c, nullptr);
      gemm(Mb, c, XbT, c, T, nullptr, c, c, c, -1.f, 2.f, 0, nullptr, nullptr); // T = 2I - MX
      trp_b(TtB, T, c, c, nullptr, 1.f, 0.f);
      gemm(Xb, c, TtB, c, X, nullptr, c, c, c, 1.f, 0.f, 0, nullptr, nullptr);  // X = X T
    }
    conv(Xb, X, (size_t)c * c, nullptr);
    trp_b(XbT, X, c, c, nullptr, 1.f, 0.f);
    trp_b(Btop, Mmat, c, c, nullptr, -1.f, 2.f);   // Btop = (I-A)^T = 2I - M^T (N x K operand)
    gemm(Xb, c, Btop, c, Qf, Qb, c, c, c, 1.f, 0.f, 0, nullptr, nullptr);       // Q[:c]=X(I-A)
    if (r > 0)
      gemm(Vb, c, XbT, c,
           Qf ? Qf + (size_t)c * c : nullptr,
           Qb ? Qb + (size_t)c * c : nullptr,
           r, c, c, -2.f, 0.f, 0, nullptr, nullptr);                            // Q[c:]=-2VX
  };

  // ---- build the four orthogonal maps ----
  cayley_Q(W_hm, NH, Md, a_hm, Qb_hm, nullptr);             // 4096 x 128
  cayley_Q(W_ux, XD, Md, a_ux, Qb_ux, nullptr);             // 4096 x 1024
  cayley_Q(W_mm, Md, Md, a_mm, Qb_mm, nullptr);             // 4096 x 4096 (skew A)
  trp_f(Wt, W_mh, NH, Md, nullptr, 1.f, 0.f);               // Wt = W_mh^T (4096 x 128)
  cayley_Q(Wt, NH, Md, a_mh, nullptr, Qt);                  // Qt = cayley(W_mh^T)
  trp_b(Bf_mh, Qt, Md, NH, nullptr, 1.f, 0.f);              // Bf_mh = Qt^T bf16 (128 x 4096)

  // ---- forward pass ----
  conv(hb, h, (size_t)Bn * NH, nullptr);
  conv(xb, x, (size_t)Bn * XD, nullptr);
  // z = relu(h @ Q_hm^T + x @ Q_ux^T + b_hm + b_ux)  (dual-K fused GEMM, bf16 out)
  gemm(hb, NH, Qb_hm, NH, nullptr, zb, Bn, Md, Md, 1.f, 0.f, 1, b_hm, b_ux,
       xb, XD, Qb_ux, XD);
  // z2 = relu(z @ Q_mm^T + b_mm)
  gemm(zb, Md, Qb_mm, Md, nullptr, z2b, Bn, Md, Md, 1.f, 0.f, 1, b_mm, nullptr);
  // f_tilde = z2 @ Qt + b_mh   (TN form with Bf_mh = Qt^T)
  gemm(z2b, Md, Bf_mh, Md, ftb, nullptr, Bn, NH, NH, 1.f, 0.f, 0, b_mh, nullptr);
  // projection onto {sum f = 0, f >= lower}
  project_k<<<Bn / 8, 256, 0, stream>>>(ftb, h, (float*)d_out, Bn);
}
